// TinyRecursiveNetwork_38508676776501
// MI455X (gfx1250) — matrix operand order repaired and checked by reading, not compiled, not run on hardware
//
#include <hip/hip_runtime.h>

// ---------------------------------------------------------------------------
// Types for CDNA5 WMMA (wave32): bf16 16x16x32, fp32 accumulate.
// ---------------------------------------------------------------------------
typedef __bf16 bf16;
typedef __attribute__((ext_vector_type(8)))  __bf16 v8bf;
typedef __attribute__((ext_vector_type(16))) __bf16 v16bf;
typedef __attribute__((ext_vector_type(8)))  float  v8f;

#define BM 128
#define BN 256
#define BK 32
#define NTHREADS 256   // 8 waves (wave32), 2x4 wave grid, 64x64 tile per wave

// gfx1250 async global->LDS copy (ASYNCcnt-tracked), with compile-safe fallback.
#if __has_builtin(__builtin_amdgcn_global_load_async_to_lds_b128)
#define HAVE_ASYNC_COPY 1
#else
#define HAVE_ASYNC_COPY 0
#endif

typedef __attribute__((address_space(1))) void gvoid_t;
typedef __attribute__((address_space(3))) void svoid_t;
typedef int v4i __attribute__((vector_size(16)));
typedef __attribute__((address_space(1))) v4i gv4i_t;
typedef __attribute__((address_space(3))) v4i sv4i_t;

__device__ __forceinline__ void async_b128(const bf16* g, bf16* l) {
#if HAVE_ASYNC_COPY
  __builtin_amdgcn_global_load_async_to_lds_b128(
      (gv4i_t*)(gvoid_t*)(void*)g,
      (sv4i_t*)(svoid_t*)(void*)l, 0, 0);
#else
  (void)g; (void)l;
#endif
}

__device__ __forceinline__ void wait_async0() {
#if HAVE_ASYNC_COPY
  asm volatile("s_wait_asynccnt 0" ::: "memory");
#endif
}

__device__ __forceinline__ v16bf cat8(v8bf lo, v8bf hi) {
  return __builtin_shufflevector(lo, hi, 0,1,2,3,4,5,6,7,8,9,10,11,12,13,14,15);
}

// ---------------------------------------------------------------------------
// Tiled bf16 GEMM:  out[M,N] = A[M,K] @ Bm[N,K]^T + bias[N]   (row-major)
// EPI 0: fp32 out
// EPI 1: bf16 out with ReLU  (feeds next GEMM directly)
// EPI 2: fp32 out (logits) + fp32 tanh(out) to second buffer (ans)
// ---------------------------------------------------------------------------
template<int EPI>
__global__ __launch_bounds__(NTHREADS)
void gemm_bf16_wmma(const bf16* __restrict__ A,
                    const bf16* __restrict__ Bm,
                    const float* __restrict__ bias,
                    float* __restrict__ outF,
                    float* __restrict__ outF2,
                    bf16*  __restrict__ outB,
                    int M, int N, int K)
{
  __shared__ bf16 sA[2][BM * BK];   // 2 x 8KB
  __shared__ bf16 sB[2][BN * BK];   // 2 x 16KB

  const int tid = threadIdx.x;
  const int m0  = blockIdx.y * BM;
  const int n0  = blockIdx.x * BN;

  // --- global -> LDS staging mapping: 64B per tile row, 4 x b128 per row ---
  const int ldr = tid >> 2;            // 0..63 (passes cover 128 / 256 rows)
  const int ldc = (tid & 3) * 8;       // element column (8 bf16 = 16B)
  const bf16* aP0 = A  + (size_t)(m0 + ldr)       * K + ldc;
  const bf16* aP1 = A  + (size_t)(m0 + 64  + ldr) * K + ldc;
  const bf16* bP0 = Bm + (size_t)(n0 + ldr)       * K + ldc;
  const bf16* bP1 = Bm + (size_t)(n0 + 64  + ldr) * K + ldc;
  const bf16* bP2 = Bm + (size_t)(n0 + 128 + ldr) * K + ldc;
  const bf16* bP3 = Bm + (size_t)(n0 + 192 + ldr) * K + ldc;

  // --- wave tiling: 2x4 waves, each wave owns a 64x64 (M x N) tile ---------
  const int lane  = tid & 31;
  const int wave  = tid >> 5;
  const int waveM = wave >> 2;         // 0..1  -> 64 rows
  const int waveN = wave & 3;          // 0..3  -> 64 cols
  const int lrow  = lane & 15;
  const int khalf = lane >> 4;

  v8f acc[4][4];
  {
    const v8f z = {0.f,0.f,0.f,0.f,0.f,0.f,0.f,0.f};
    #pragma unroll
    for (int mi = 0; mi < 4; ++mi)
      #pragma unroll
      for (int ni = 0; ni < 4; ++ni)
        acc[mi][ni] = z;
  }

  const int nk = K / BK;

#if HAVE_ASYNC_COPY
  // -------- async GLOBAL_LOAD_ASYNC_TO_LDS_B128 pipeline -------------------
  {
    async_b128(aP0, &sA[0][ldr        * BK + ldc]);
    async_b128(aP1, &sA[0][(64  + ldr) * BK + ldc]);
    async_b128(bP0, &sB[0][ldr        * BK + ldc]);
    async_b128(bP1, &sB[0][(64  + ldr) * BK + ldc]);
    async_b128(bP2, &sB[0][(128 + ldr) * BK + ldc]);
    async_b128(bP3, &sB[0][(192 + ldr) * BK + ldc]);
  }
  wait_async0();
  __syncthreads();
#else
  // -------- fallback: register-staged loads + ds_store ---------------------
  uint4 ra0 = *(const uint4*)aP0;
  uint4 ra1 = *(const uint4*)aP1;
  uint4 rb0 = *(const uint4*)bP0;
  uint4 rb1 = *(const uint4*)bP1;
  uint4 rb2 = *(const uint4*)bP2;
  uint4 rb3 = *(const uint4*)bP3;
  *(uint4*)&sA[0][ldr        * BK + ldc] = ra0;
  *(uint4*)&sA[0][(64  + ldr) * BK + ldc] = ra1;
  *(uint4*)&sB[0][ldr        * BK + ldc] = rb0;
  *(uint4*)&sB[0][(64  + ldr) * BK + ldc] = rb1;
  *(uint4*)&sB[0][(128 + ldr) * BK + ldc] = rb2;
  *(uint4*)&sB[0][(192 + ldr) * BK + ldc] = rb3;
  __syncthreads();
#endif

  for (int kt = 0; kt < nk; ++kt) {
    const int buf = kt & 1;
    const bool hasNext = (kt + 1) < nk;
    const int nb = buf ^ 1;
    const size_t go = (size_t)(kt + 1) * BK;

#if HAVE_ASYNC_COPY
    if (hasNext) {
      // safe: nb was last read at iteration kt-1, sealed by that barrier
      async_b128(aP0 + go, &sA[nb][ldr        * BK + ldc]);
      async_b128(aP1 + go, &sA[nb][(64  + ldr) * BK + ldc]);
      async_b128(bP0 + go, &sB[nb][ldr        * BK + ldc]);
      async_b128(bP1 + go, &sB[nb][(64  + ldr) * BK + ldc]);
      async_b128(bP2 + go, &sB[nb][(128 + ldr) * BK + ldc]);
      async_b128(bP3 + go, &sB[nb][(192 + ldr) * BK + ldc]);
    }
#else
    if (hasNext) {
      ra0 = *(const uint4*)(aP0 + go);
      ra1 = *(const uint4*)(aP1 + go);
      rb0 = *(const uint4*)(bP0 + go);
      rb1 = *(const uint4*)(bP1 + go);
      rb2 = *(const uint4*)(bP2 + go);
      rb3 = *(const uint4*)(bP3 + go);
    }
#endif

    // Fragment fills per ISA 16-bit layouts:
    //  A 16x32: lane l holds row (l&15); K = khalf*8 + [0..7] and 16+khalf*8 + [0..7]
    //  B 32x16: lane l holds col (l&15); K = khalf*8 + [0..7] and 16+khalf*8 + [0..7]
    v16bf af[4];
    #pragma unroll
    for (int mi = 0; mi < 4; ++mi) {
      const bf16* p = &sA[buf][(waveM*64 + mi*16 + lrow) * BK + khalf*8];
      af[mi] = cat8(*(const v8bf*)p, *(const v8bf*)(p + 16));
    }
    #pragma unroll
    for (int ni = 0; ni < 4; ++ni) {
      const bf16* p = &sB[buf][(waveN*64 + ni*16 + lrow) * BK + khalf*8];
      const v16bf bb = cat8(*(const v8bf*)p, *(const v8bf*)(p + 16));
      #pragma unroll
      for (int mi = 0; mi < 4; ++mi)
        acc[mi][ni] = __builtin_amdgcn_wmma_f32_16x16x32_bf16(
            false, af[mi], false, bb, (short)0, acc[mi][ni], false, false);
    }

    if (hasNext) {
#if HAVE_ASYNC_COPY
      wait_async0();
      __syncthreads();
#else
      __syncthreads();
      *(uint4*)&sA[nb][ldr        * BK + ldc] = ra0;
      *(uint4*)&sA[nb][(64  + ldr) * BK + ldc] = ra1;
      *(uint4*)&sB[nb][ldr        * BK + ldc] = rb0;
      *(uint4*)&sB[nb][(64  + ldr) * BK + ldc] = rb1;
      *(uint4*)&sB[nb][(128 + ldr) * BK + ldc] = rb2;
      *(uint4*)&sB[nb][(192 + ldr) * BK + ldc] = rb3;
      __syncthreads();
#endif
    }
  }

  // --- epilogue: C/D layout -> lane (l&15) = N, VGPR v = row (v + (l>>4)*8)
  #pragma unroll
  for (int mi = 0; mi < 4; ++mi) {
    #pragma unroll
    for (int ni = 0; ni < 4; ++ni) {
      const int n  = n0 + waveN*64 + ni*16 + lrow;
      const float bv = bias ? bias[n] : 0.0f;
      const int mb = m0 + waveM*64 + mi*16 + khalf*8;
      #pragma unroll
      for (int v = 0; v < 8; ++v) {
        const float val = acc[mi][ni][v] + bv;
        const size_t off = (size_t)(mb + v) * N + n;
        if (EPI == 0) {
          outF[off] = val;
        } else if (EPI == 1) {
          outB[off] = (bf16)fmaxf(val, 0.0f);
        } else {
          outF[off]  = val;
          outF2[off] = tanhf(val);
        }
      }
    }
  }
}

// ---------------------------------------------------------------------------
// Elementwise helpers
// ---------------------------------------------------------------------------
__global__ void cvt_f32_bf16_kernel(bf16* __restrict__ dst,
                                    const float* __restrict__ src,
                                    long long n) {
  long long i = (long long)blockIdx.x * blockDim.x + threadIdx.x;
  if (i < n) dst[i] = (bf16)src[i];
}

__global__ void zero_f32_kernel(float* __restrict__ dst, long long n) {
  long long i = (long long)blockIdx.x * blockDim.x + threadIdx.x;
  if (i < n) dst[i] = 0.0f;
}

// dst[r, 0:ca) = bf16(a[r]);  dst[r, ca:ca+cb) = bf16(b[r])
__global__ void concat2_bf16_kernel(bf16* __restrict__ dst,
                                    const float* __restrict__ a, int ca,
                                    const float* __restrict__ b, int cb,
                                    int rows) {
  const int cols = ca + cb;
  long long i = (long long)blockIdx.x * blockDim.x + threadIdx.x;
  if (i >= (long long)rows * cols) return;
  const int r = (int)(i / cols);
  const int c = (int)(i % cols);
  const float v = (c < ca) ? a[(size_t)r * ca + c] : b[(size_t)r * cb + (c - ca)];
  dst[i] = (bf16)v;
}

// GRU gate fusion: state = (1-z)*n + z*state,  gates ordered (r, z, n)
__global__ void gru_gate_kernel(float* __restrict__ state,
                                const float* __restrict__ xg,
                                const float* __restrict__ hg,
                                int B, int L) {
  long long i = (long long)blockIdx.x * blockDim.x + threadIdx.x;
  if (i >= (long long)B * L) return;
  const int r = (int)(i / L);
  const int c = (int)(i % L);
  const size_t base = (size_t)r * (3 * L) + c;
  const float xr = xg[base], xz = xg[base + L], xn = xg[base + 2*L];
  const float hr = hg[base], hz = hg[base + L], hn = hg[base + 2*L];
  const float rg = 1.0f / (1.0f + __expf(-(xr + hr)));
  const float zg = 1.0f / (1.0f + __expf(-(xz + hz)));
  const float ng = tanhf(xn + rg * hn);
  const float s  = state[i];
  state[i] = (1.0f - zg) * ng + zg * s;
}

// halt = h2 @ hw2^T + hb2   (N=2, K=HH): one wave per row, lane-strided dot
__global__ void halt_kernel(float* __restrict__ out,
                            const bf16* __restrict__ h2,
                            const float* __restrict__ w,
                            const float* __restrict__ b,
                            int B, int H) {
  const int row  = blockIdx.x * (blockDim.x >> 5) + (threadIdx.x >> 5);
  const int lane = threadIdx.x & 31;
  if (row >= B) return;
  float s0 = 0.0f, s1 = 0.0f;
  for (int k = lane; k < H; k += 32) {
    const float h = (float)h2[(size_t)row * H + k];
    s0 += h * w[k];
    s1 += h * w[H + k];
  }
  #pragma unroll
  for (int o = 16; o > 0; o >>= 1) {
    s0 += __shfl_down(s0, o, 32);
    s1 += __shfl_down(s1, o, 32);
  }
  if (lane == 0) {
    out[(size_t)row * 2 + 0] = s0 + b[0];
    out[(size_t)row * 2 + 1] = s1 + b[1];
  }
}

// ---------------------------------------------------------------------------
// Host orchestration
// ---------------------------------------------------------------------------
extern "C" void kernel_launch(void* const* d_in, const int* in_sizes, int n_in,
                              void* d_out, int out_size, void* d_ws, size_t ws_size,
                              hipStream_t stream) {
  (void)in_sizes; (void)n_in; (void)out_size; (void)ws_size;

  constexpr int Bb = 4096, I = 1024, L = 1024, H = 2048, O = 512, HH = 1024;
  constexpr int GIN = I + O;          // 1536
  constexpr int JD  = L + O;          // 1536
  constexpr int G3  = 3 * L;          // 3072
  constexpr int STEPS = 8, INNER = 4; // reference module constants

  const float* inputs = (const float*)d_in[0];
  const float* W_ih   = (const float*)d_in[1];
  const float* W_hh   = (const float*)d_in[2];
  const float* b_ih   = (const float*)d_in[3];
  const float* b_hh   = (const float*)d_in[4];
  const float* aw1    = (const float*)d_in[5];
  const float* ab1    = (const float*)d_in[6];
  const float* aw2    = (const float*)d_in[7];
  const float* ab2    = (const float*)d_in[8];
  const float* hw1    = (const float*)d_in[9];
  const float* hb1    = (const float*)d_in[10];
  const float* hw2    = (const float*)d_in[11];
  const float* hb2    = (const float*)d_in[12];

  // workspace bump allocator (256B aligned)
  char* wsb = (char*)d_ws;
  size_t off = 0;
  auto alloc = [&](size_t bytes) -> void* {
    void* p = wsb + off;
    off = (off + bytes + 255) & ~(size_t)255;
    return p;
  };
  bf16* bWih   = (bf16*)alloc((size_t)G3 * GIN * sizeof(bf16));
  bf16* bWhh   = (bf16*)alloc((size_t)G3 * L   * sizeof(bf16));
  bf16* bAw1   = (bf16*)alloc((size_t)H  * JD  * sizeof(bf16));
  bf16* bAw2   = (bf16*)alloc((size_t)O  * H   * sizeof(bf16));
  bf16* bHw1   = (bf16*)alloc((size_t)HH * JD  * sizeof(bf16));
  bf16* xcat   = (bf16*)alloc((size_t)Bb * GIN * sizeof(bf16));
  bf16* joint  = (bf16*)alloc((size_t)Bb * JD  * sizeof(bf16));
  bf16* stateb = (bf16*)alloc((size_t)Bb * L   * sizeof(bf16));
  bf16* h1b    = (bf16*)alloc((size_t)Bb * H   * sizeof(bf16));
  bf16* h2b    = (bf16*)alloc((size_t)Bb * HH  * sizeof(bf16));
  float* xg    = (float*)alloc((size_t)Bb * G3 * sizeof(float));
  float* hg    = (float*)alloc((size_t)Bb * G3 * sizeof(float));
  float* state = (float*)alloc((size_t)Bb * L  * sizeof(float));
  float* ans   = (float*)alloc((size_t)Bb * O  * sizeof(float));

  const dim3 blk(NTHREADS);
  auto cvt = [&](bf16* dst, const float* src, long long n) {
    cvt_f32_bf16_kernel<<<(unsigned)((n + 255) / 256), 256, 0, stream>>>(dst, src, n);
  };
  auto zero = [&](float* dst, long long n) {
    zero_f32_kernel<<<(unsigned)((n + 255) / 256), 256, 0, stream>>>(dst, n);
  };

  // One-time (per call) weight conversion fp32 -> bf16
  cvt(bWih, W_ih, (long long)G3 * GIN);
  cvt(bWhh, W_hh, (long long)G3 * L);
  cvt(bAw1, aw1, (long long)H * JD);
  cvt(bAw2, aw2, (long long)O * H);
  cvt(bHw1, hw1, (long long)HH * JD);
  zero(state, (long long)Bb * L);
  zero(ans,   (long long)Bb * O);

  float* outLogits = (float*)d_out;                              // [8, B, O]
  float* outHalt   = (float*)d_out + (size_t)STEPS * Bb * O;     // [8, B, 2]

  for (int t = 0; t < STEPS; ++t) {
    // x = concat(inputs, ans) -> bf16; x_gates = x @ W_ih^T + b_ih
    concat2_bf16_kernel<<<(unsigned)(((long long)Bb * GIN + 255) / 256), 256, 0, stream>>>(
        xcat, inputs, I, ans, O, Bb);
    gemm_bf16_wmma<0><<<dim3(G3 / BN, Bb / BM), blk, 0, stream>>>(
        xcat, bWih, b_ih, xg, nullptr, nullptr, Bb, G3, GIN);

    // inner GRU cycles
    for (int ic = 0; ic < INNER; ++ic) {
      cvt(stateb, state, (long long)Bb * L);
      gemm_bf16_wmma<0><<<dim3(G3 / BN, Bb / BM), blk, 0, stream>>>(
          stateb, bWhh, b_hh, hg, nullptr, nullptr, Bb, G3, L);
      gru_gate_kernel<<<(unsigned)(((long long)Bb * L + 255) / 256), 256, 0, stream>>>(
          state, xg, hg, Bb, L);
    }

    // joint = concat(state, ans) -> bf16
    concat2_bf16_kernel<<<(unsigned)(((long long)Bb * JD + 255) / 256), 256, 0, stream>>>(
        joint, state, L, ans, O, Bb);

    // h1 = relu(joint @ aw1^T + ab1) -> bf16
    gemm_bf16_wmma<1><<<dim3(H / BN, Bb / BM), blk, 0, stream>>>(
        joint, bAw1, ab1, nullptr, nullptr, h1b, Bb, H, JD);

    // logits = h1 @ aw2^T + ab2 -> d_out slice;  ans = tanh(logits)
    gemm_bf16_wmma<2><<<dim3(O / BN, Bb / BM), blk, 0, stream>>>(
        h1b, bAw2, ab2, outLogits + (size_t)t * Bb * O, ans, nullptr, Bb, O, H);

    // h2 = relu(joint @ hw1^T + hb1) -> bf16
    gemm_bf16_wmma<1><<<dim3(HH / BN, Bb / BM), blk, 0, stream>>>(
        joint, bHw1, hb1, nullptr, nullptr, h2b, Bb, HH, JD);

    // halt = h2 @ hw2^T + hb2  (N=2 reduction kernel, negligible FLOPs)
    halt_kernel<<<(unsigned)((Bb + 7) / 8), 256, 0, stream>>>(
        outHalt + (size_t)t * Bb * 2, h2b, hw2, hb2, Bb, HH);
  }
}
